// RNN_63737314673219
// MI455X (gfx1250) — matrix operand order repaired and checked by reading, not compiled, not run on hardware
//
#include <hip/hip_runtime.h>
#include <hip/hip_bf16.h>

// Bidirectional LSTM for MI455X (gfx1250), bf16 WMMA with fp32 accumulate.
//
//   - prep kernels: x -> bf16 time-major [T,B,D]; W -> bf16 transposed [4H,K],
//     K contiguous per column; swizzled at LDS preload.
//   - persistent kernel: 64 WGs (2 dirs x 32 h-column chunks), 4 waves each.
//     Each WG holds its 1024x64 bf16 weight slice in LDS (128 KB of the
//     320 KB/WGP pool) swizzled as [kk][col][32] so all ds_loads in an
//     unrolled group use 16-bit immediate offsets. Cell state c stays in
//     VGPRs; a device-wide atomic-counter barrier separates time steps.
//   - per step per wave: 4 gates x 32 K-steps of v_wmma_f32_16x16x32_bf16.

typedef __attribute__((ext_vector_type(16))) __bf16 v16bf;
typedef __attribute__((ext_vector_type(8)))  float  v8f;

#define B_   64
#define T_   512
#define D_   512
#define H_   512
#define NG_  2048   // 4*H
#define KT_  1024   // D+H
#define NWG  64     // persistent grid: 2 dirs * 32 column chunks
#define NTHR 128    // 4 waves (wave32)

union ABFrag { v16bf v; uint4 q[2]; };

__device__ __forceinline__ float fsig_(float x) {
  return __fdividef(1.0f, 1.0f + __expf(-x));
}
__device__ __forceinline__ float ftanh_(float x) {
  float e = __expf(-2.0f * x);
  return __fdividef(1.0f - e, 1.0f + e);
}

// ---------------- prep kernels ------------------------------------------

// x [B,T,D] f32  ->  xbf [T,B,D] bf16 (coalesced writes)
__global__ void cvt_x_kernel(const float* __restrict__ x, __bf16* __restrict__ xbf) {
  size_t e = (size_t)blockIdx.x * blockDim.x + threadIdx.x;   // T*B*D threads
  int k  = (int)(e & (D_ - 1));
  size_t bt = e >> 9;              // D_=512
  int b = (int)(bt & (B_ - 1));    // B_=64
  int t = (int)(bt >> 6);
  xbf[e] = (__bf16)x[((size_t)b * T_ + t) * D_ + k];
}

// W [K,4H] f32 -> wT [4H,K] bf16 (column-major in K: K contiguous per column)
__global__ void cvt_w_kernel(const float* __restrict__ W, __bf16* __restrict__ wT) {
  size_t e = (size_t)blockIdx.x * blockDim.x + threadIdx.x;   // NG*KT threads
  int n = (int)(e >> 10);          // KT_=1024
  int k = (int)(e & (KT_ - 1));
  wT[e] = (__bf16)W[(size_t)k * NG_ + n];
}

// Reset barrier counter and zero the h0 buffers (both directions).
// hbuf layout: [dir*2 + buf][B_*H_] bf16; buf0 of dir d at uint4 offset d*8192.
__global__ void init_ws_kernel(unsigned* __restrict__ cnt, uint4* __restrict__ hbuf4) {
  size_t e = (size_t)blockIdx.x * blockDim.x + threadIdx.x;   // 8192 threads
  if (e == 0) *cnt = 0u;
  uint4 z; z.x = z.y = z.z = z.w = 0u;
  size_t o = (e < 4096) ? e : (e - 4096 + 8192);
  hbuf4[o] = z;
}

// ---------------- device-wide barrier -----------------------------------

__device__ __forceinline__ void gsync(unsigned* cnt, unsigned target) {
  __threadfence();          // release our h stores device-wide
  __syncthreads();
  if (threadIdx.x == 0) {
    atomicAdd(cnt, 1u);
    volatile unsigned* vc = cnt;
    while (*vc < target) __builtin_amdgcn_s_sleep(2);
  }
  __syncthreads();
  __threadfence();          // acquire: invalidate stale near caches
}

// ---------------- persistent recurrent kernel ---------------------------

__global__ void __launch_bounds__(NTHR, 1)
lstm_persist_kernel(const __bf16* __restrict__ xbf,   // [T,B,D]
                    const __bf16* __restrict__ wT,    // [dir][4H][K] col-major in K
                    const float*  __restrict__ b_fw,
                    const float*  __restrict__ b_bw,
                    __bf16* __restrict__ hbuf,        // [dir][2][B,H] bf16
                    float*  __restrict__ out,         // [B,T,2H] f32
                    unsigned* __restrict__ cnt) {
  extern __shared__ __align__(16) __bf16 sW[];        // 128 KB, [kk][col][32]

  const int tid  = threadIdx.x;
  const int wv   = tid >> 5;          // M-tile (batch group of 16)
  const int lane = tid & 31;
  const int dir  = blockIdx.x >> 5;   // 0 = fw, 1 = bw
  const int n0   = (blockIdx.x & 31) << 4;   // base h-column of this WG

  // ---- preload 1024x64 bf16 weight slice into LDS, K-block swizzled ----
  // LDS element layout: bf16 index = (kk*64 + col)*32 + klocal,
  //   kk = K/32 block (0..31), col = gate*16 + j (0..63), klocal = K%32.
  {
    const uint4* wsrc = (const uint4*)(wT + (size_t)dir * NG_ * KT_);  // 128 uint4/col
    uint4* sW4 = (uint4*)sW;
    for (int idx = tid; idx < 64 * 128; idx += NTHR) {
      int c = idx >> 7;           // col 0..63
      int q = idx & 127;          // 16B chunk within column (k = q*8)
      int n = ((c >> 4) * H_) + n0 + (c & 15);          // global column in [0,4H)
      int kk = q >> 2, kl = q & 3;
      sW4[(kk * 64 + c) * 4 + kl] = wsrc[(size_t)n * 128 + q];
    }
  }
  __syncthreads();

  const float* bvec = dir ? b_bw : b_fw;
  const int ncol = n0 + (lane & 15);                    // this lane's h column
  float bias[4];
#pragma unroll
  for (int g = 0; g < 4; ++g) bias[g] = bvec[g * H_ + ncol];

  // A fragment addressing (16-bit A 16x32 layout): lane m = lane&15,
  // K chunks [kb,kb+8) and [kb+16,kb+24) with kb = (lane>>4)*8.
  const int m   = lane & 15;
  const int kb  = (lane >> 4) << 3;
  // B fragment (32x16): lane n = lane&15 holds K [kb,kb+8) and [kb+16,kb+24), same as A.
  const int kbB = (lane >> 4) << 4;
  const int nB  = lane & 15;
  const int mb  = (lane >> 4) << 3;   // C/D layout row base
  const unsigned bOff0 = (unsigned)(nB * 64 + kb * 2); // lane byte off in kk-block
  const char* sWc = (const char*)sW;

  v8f cst = {};                        // cell-state tile, fp32, lives in VGPRs

  for (int t = 0; t < T_; ++t) {
    const int tt = dir ? (T_ - 1 - t) : t;
    const __bf16* xrow = xbf + (size_t)tt * (B_ * D_) + (size_t)(wv * 16 + m) * D_ + kb;
    const __bf16* hrow = hbuf + (size_t)(dir * 2 + (t & 1)) * (B_ * H_)
                              + (size_t)(wv * 16 + m) * H_ + kb;
    __bf16* hw = hbuf + (size_t)(dir * 2 + ((t + 1) & 1)) * (B_ * H_);

    v8f acc[4];
#pragma unroll
    for (int g = 0; g < 4; ++g) {
      v8f a;
#pragma unroll
      for (int r = 0; r < 8; ++r) a[r] = bias[g];
      acc[g] = a;
    }

    // z = [x_t | h] @ W : K = 1024 = 32 WMMA K-steps of 32.
    // Phase 0 streams x_t rows, phase 1 streams h rows (no per-iter select).
    unsigned bOff = bOff0;
#pragma unroll
    for (int ph = 0; ph < 2; ++ph) {
      const __bf16* arow = (ph == 0) ? xrow : hrow;
#pragma unroll 4
      for (int kk = 0; kk < 16; ++kk) {
        ABFrag a;
        const uint4* apq = (const uint4*)(arow + kk * 32);
        a.q[0] = apq[0];   // K kb..kb+7
        a.q[1] = apq[2];   // K kb+16..kb+23
#pragma unroll
        for (int g = 0; g < 4; ++g) {
          ABFrag bf;
          const uint4* bp = (const uint4*)(sWc + bOff + g * 1024);
          bf.q[0] = bp[0];
          bf.q[1] = bp[2];
          acc[g] = __builtin_amdgcn_wmma_f32_16x16x32_bf16(
              false, a.v, false, bf.v, (short)0, acc[g], false, false);
        }
        bOff += 4096;      // next kk-block; immediates cover each unrolled group
      }
    }

    // gates: i,j,f,o ; c = sig(f+1)*c + sig(i)*tanh(j) ; h = sig(o)*tanh(c)
#pragma unroll
    for (int r = 0; r < 8; ++r) {
      float iv = fsig_(acc[0][r]);
      float jv = ftanh_(acc[1][r]);
      float fv = fsig_(acc[2][r] + 1.0f);
      float ov = fsig_(acc[3][r]);
      float cv = fv * cst[r] + iv * jv;
      cst[r] = cv;
      float hv = ov * ftanh_(cv);
      int b = wv * 16 + mb + r;
      hw[(size_t)b * H_ + ncol] = (__bf16)hv;
      out[((size_t)b * T_ + tt) * (2 * H_) + (size_t)dir * H_ + ncol] = hv;
    }

    if (t != T_ - 1) {
      // prefetch next step's x rows (global_prefetch_b8) to hide latency
      // behind the barrier on the serial critical path
      const int tn = dir ? (T_ - 2 - t) : (t + 1);
      const char* pf = (const char*)(xbf + (size_t)tn * (B_ * D_)
                                         + (size_t)(wv * 16 + m) * D_);
      const int half = (lane >> 4) * 512;
      __builtin_prefetch(pf + half, 0, 1);
      __builtin_prefetch(pf + half + 256, 0, 1);
      gsync(cnt, (unsigned)(t + 1) * NWG);
    }
  }
}

// ---------------- host launcher ------------------------------------------

extern "C" void kernel_launch(void* const* d_in, const int* in_sizes, int n_in,
                              void* d_out, int out_size, void* d_ws, size_t ws_size,
                              hipStream_t stream) {
  (void)in_sizes; (void)n_in; (void)out_size; (void)ws_size;
  const float* x    = (const float*)d_in[0];
  const float* W_fw = (const float*)d_in[1];
  const float* b_fw = (const float*)d_in[2];
  const float* W_bw = (const float*)d_in[3];
  const float* b_bw = (const float*)d_in[4];
  float* out = (float*)d_out;

  char* ws = (char*)d_ws;
  unsigned* cnt = (unsigned*)ws;                                   // 256 B reserved
  __bf16* hbuf  = (__bf16*)(ws + 256);                             // 256 KB
  __bf16* xbf   = (__bf16*)(ws + 256 + 262144);                    // 32 MB
  __bf16* wT    = (__bf16*)(ws + 256 + 262144 + (size_t)T_ * B_ * D_ * 2);  // 8 MB

  cvt_x_kernel<<<(T_ * B_ * D_) / 256, 256, 0, stream>>>(x, xbf);
  cvt_w_kernel<<<(NG_ * KT_) / 256, 256, 0, stream>>>(W_fw, wT);
  cvt_w_kernel<<<(NG_ * KT_) / 256, 256, 0, stream>>>(W_bw, wT + (size_t)NG_ * KT_);
  init_ws_kernel<<<32, 256, 0, stream>>>(cnt, (uint4*)hbuf);

  lstm_persist_kernel<<<NWG, NTHR, 64 * 1024 * (int)sizeof(__bf16), stream>>>(
      xbf, wT, b_fw, b_bw, hbuf, out, cnt);
}
